// TokenRoutedMLPParallel_76209899700388
// MI455X (gfx1250) — matrix operand order repaired and checked by reading, not compiled, not run on hardware
//
#include <hip/hip_runtime.h>
#include <cstdint>

#define T_TOK   16384
#define D_HID   1024
#define HE      512
#define NE      8
#define TM      32
#define XSTR    1032                 // halfs per X row in LDS (1024 + 8 pad)
#define ISTR    520                  // halfs per inter row in LDS (512 + 8 pad)
#define X_BYTES (TM * XSTR * 2)      // 66048
#define I_BYTES (TM * ISTR * 2)      // 33280
#define SMEM_BYTES (X_BYTES + I_BYTES + TM * 4)  // 99456
#define VOCAB   50257

typedef __attribute__((ext_vector_type(16))) __bf16 v16bf;
typedef __attribute__((ext_vector_type(8)))  float  v8f;

union BF16Frag { v16bf v; uint4 q[2]; };

__device__ __forceinline__ uint16_t f2bf(float f) {
  union { float f; uint32_t u; } c; c.f = f;
  const uint32_t u = c.u;
  return (uint16_t)((u + 0x7FFFu + ((u >> 16) & 1u)) >> 16);
}

__device__ __forceinline__ v8f vzero() {
  v8f z = {0.f, 0.f, 0.f, 0.f, 0.f, 0.f, 0.f, 0.f};
  return z;
}

// A fragment (16x32 bf16, M x K) from LDS, contiguous k order (k = 16h + s), same as B:
// lanes 0-15: row m0+lane, K = k0..k0+15
// lanes 16-31: row m0+lane-16, K = k0+16..k0+31
__device__ __forceinline__ v16bf load_a_lds(const uint16_t* row0, int strideH,
                                            int m0, int k0, int lane) {
  const int row  = m0 + (lane & 15);
  const int koff = k0 + ((lane >> 4) << 4);
  const uint16_t* p = row0 + row * strideH + koff;
  BF16Frag f;
  f.q[0] = *(const uint4*)(p);       // 8 halfs
  f.q[1] = *(const uint4*)(p + 8);   // 8 halfs
  return f.v;
}

// B fragment (32x16 bf16, K x N) from global weights stored [N][K]:
// lanes 0-15: col n0+lane, K = k0..k0+15 ; lanes 16-31: col n0+lane-16, K = k0+16..k0+31
__device__ __forceinline__ v16bf load_b_glb(const uint16_t* w, int Kdim,
                                            int n0, int k0, int lane) {
  const uint16_t* p = w + (size_t)(n0 + (lane & 15)) * Kdim + k0 + ((lane >> 4) << 4);
  BF16Frag f;
  f.q[0] = ((const uint4*)p)[0];
  f.q[1] = ((const uint4*)p)[1];
  return f.v;
}

__device__ __forceinline__ v8f wmma_bf16(v16bf a, v16bf b, v8f c) {
  return __builtin_amdgcn_wmma_f32_16x16x32_bf16(false, a, false, b, (short)0, c,
                                                 false, false);
}

__global__ void init_counts(int* __restrict__ counts) {
  if (threadIdx.x < NE) counts[threadIdx.x] = 0;
}

__global__ __launch_bounds__(256) void route_kernel(
    const float* __restrict__ mu, const int* __restrict__ token_ids,
    const float* __restrict__ rw, int* __restrict__ counts, int* __restrict__ lists) {
  __shared__ float wsh[NE * D_HID];
  const int tid = threadIdx.x;
  for (int i = tid; i < NE * D_HID; i += 256) wsh[i] = rw[i];
  __syncthreads();
  const int wave = tid >> 5, lane = tid & 31;
  const int t = blockIdx.x * 8 + wave;
  float acc[NE];
#pragma unroll
  for (int e = 0; e < NE; ++e) acc[e] = 0.f;
  const float* mrow = mu + (size_t)t * D_HID;
  for (int i = 0; i < D_HID; i += 32) {
    const float mv = mrow[i + lane];
#pragma unroll
    for (int e = 0; e < NE; ++e) acc[e] += mv * wsh[e * D_HID + i + lane];
  }
#pragma unroll
  for (int e = 0; e < NE; ++e) {
    for (int off = 16; off > 0; off >>= 1) acc[e] += __shfl_xor(acc[e], off, 32);
  }
  if (lane == 0) {
    int tok = token_ids[t];
    if (tok < 0) tok = 0;
    if (tok > VOCAB - 1) tok = VOCAB - 1;
    const int base = tok & (NE - 1);
    float best = acc[0] + (base == 0 ? 10.f : 0.f);
    int bi = 0;
#pragma unroll
    for (int e = 1; e < NE; ++e) {
      const float v = acc[e] + (base == e ? 10.f : 0.f);
      if (v > best) { best = v; bi = e; }
    }
    const int pos = atomicAdd(&counts[bi], 1);
    lists[bi * T_TOK + pos] = t;
  }
}

// src: [NE][K][N] f32  ->  dst: [NE][N][K] bf16 (so B frags are K-contiguous)
__global__ __launch_bounds__(256) void transpose_cvt(
    const float* __restrict__ src, uint16_t* __restrict__ dst, int K, int N) {
  __shared__ float tile[32][33];
  const int nTilesN = N >> 5, nTilesK = K >> 5;
  int b = blockIdx.x;
  const int e = b / (nTilesK * nTilesN);
  b -= e * (nTilesK * nTilesN);
  const int kt = b / nTilesN;
  const int nt = b - kt * nTilesN;
  const float* s = src + (size_t)e * K * N + (size_t)(kt << 5) * N + (nt << 5);
  uint16_t*    d = dst + (size_t)e * N * K + (size_t)(nt << 5) * K + (kt << 5);
  const int tid = threadIdx.x;
  for (int i = tid; i < 1024; i += 256) {
    const int kk = i >> 5, nn = i & 31;
    tile[kk][nn] = s[(size_t)kk * N + nn];
  }
  __syncthreads();
  for (int i = tid; i < 1024; i += 256) {
    const int nn = i >> 5, kk = i & 31;
    d[(size_t)nn * K + kk] = f2bf(tile[kk][nn]);
  }
}

__global__ __launch_bounds__(256) void moe_kernel(
    const float* __restrict__ x, const int* __restrict__ counts,
    const int* __restrict__ lists, const uint16_t* __restrict__ gate_bf,
    const uint16_t* __restrict__ up_bf, const uint16_t* __restrict__ down_bf,
    float* __restrict__ out) {
  extern __shared__ char smem[];
  uint16_t* xs   = (uint16_t*)smem;
  uint16_t* is_  = (uint16_t*)(smem + X_BYTES);
  int*      tokT = (int*)(smem + X_BYTES + I_BYTES);

  const int tilesPerE = T_TOK / TM;
  const int e    = blockIdx.x / tilesPerE;
  const int tile = blockIdx.x - e * tilesPerE;
  const int cnt  = counts[e];
  if (tile * TM >= cnt) return;  // block-uniform early exit

  const int tid = threadIdx.x, wave = tid >> 5, lane = tid & 31;

  if (tid < TM) {
    const int idx = tile * TM + tid;
    tokT[tid] = (idx < cnt) ? lists[e * T_TOK + idx] : -1;
  }
  __syncthreads();

  // ---- stage gathered X rows into LDS as bf16 ----
  for (int i = tid; i < TM * (D_HID / 4); i += 256) {
    const int row = i >> 8, c4 = i & 255;
    int t = tokT[row]; if (t < 0) t = 0;
    const float4 v = ((const float4*)(x + (size_t)t * D_HID))[c4];
    uint32_t* d = (uint32_t*)(xs + row * XSTR + (c4 << 2));
    d[0] = (uint32_t)f2bf(v.x) | ((uint32_t)f2bf(v.y) << 16);
    d[1] = (uint32_t)f2bf(v.z) | ((uint32_t)f2bf(v.w) << 16);
  }
  __syncthreads();

  const uint16_t* gw = gate_bf + (size_t)e * D_HID * HE;
  const uint16_t* uw = up_bf   + (size_t)e * D_HID * HE;
  const uint16_t* dw = down_bf + (size_t)e * HE * D_HID;

  // ---- gate/up GEMMs: each wave owns 4 N-tiles of HE=512 ----
#pragma unroll
  for (int miter = 0; miter < 2; ++miter) {
    const int m0 = miter << 4;
    v8f g[4], u[4];
#pragma unroll
    for (int j = 0; j < 4; ++j) { g[j] = vzero(); u[j] = vzero(); }
    for (int k0 = 0; k0 < D_HID; k0 += 32) {
      const v16bf a = load_a_lds(xs, XSTR, m0, k0, lane);
#pragma unroll
      for (int j = 0; j < 4; ++j) {
        const int n0 = ((wave << 2) + j) << 4;
        g[j] = wmma_bf16(a, load_b_glb(gw, D_HID, n0, k0, lane), g[j]);
        u[j] = wmma_bf16(a, load_b_glb(uw, D_HID, n0, k0, lane), u[j]);
      }
    }
    // silu(g) * u -> bf16 intermediate in LDS (C-layout: VGPR i => row m0+i(+8 hi))
#pragma unroll
    for (int j = 0; j < 4; ++j) {
      const int n  = (((wave << 2) + j) << 4) + (lane & 15);
      const int mb = m0 + ((lane >> 4) << 3);
#pragma unroll
      for (int i = 0; i < 8; ++i) {
        const float gv = g[j][i];
        const float hv = (gv / (1.f + __expf(-gv))) * u[j][i];
        is_[(mb + i) * ISTR + n] = f2bf(hv);
      }
    }
  }
  __syncthreads();

  // ---- down GEMM: each wave owns 8 N-tiles of D=1024 ----
#pragma unroll
  for (int miter = 0; miter < 2; ++miter) {
    const int m0 = miter << 4;
    v8f y[8];
#pragma unroll
    for (int j = 0; j < 8; ++j) y[j] = vzero();
    for (int k0 = 0; k0 < HE; k0 += 32) {
      const v16bf a = load_a_lds(is_, ISTR, m0, k0, lane);
#pragma unroll
      for (int j = 0; j < 8; ++j) {
        const int n0 = ((wave << 3) + j) << 4;
        y[j] = wmma_bf16(a, load_b_glb(dw, HE, n0, k0, lane), y[j]);
      }
    }
    const int mb = m0 + ((lane >> 4) << 3);
#pragma unroll
    for (int j = 0; j < 8; ++j) {
      const int n = (((wave << 3) + j) << 4) + (lane & 15);
#pragma unroll
      for (int i = 0; i < 8; ++i) {
        const int t = tokT[mb + i];
        if (t >= 0) out[(size_t)t * D_HID + n] = y[j][i];
      }
    }
  }
}

extern "C" void kernel_launch(void* const* d_in, const int* in_sizes, int n_in,
                              void* d_out, int out_size, void* d_ws, size_t ws_size,
                              hipStream_t stream) {
  const float* hidden = (const float*)d_in[0];
  const int*   tok    = (const int*)d_in[1];
  const float* mu     = (const float*)d_in[2];
  const float* gate   = (const float*)d_in[3];
  const float* up     = (const float*)d_in[4];
  const float* down   = (const float*)d_in[5];
  const float* rw     = (const float*)d_in[6];
  float* out = (float*)d_out;

  char* ws = (char*)d_ws;
  int* counts = (int*)ws;
  int* lists  = (int*)(ws + 256);
  uint16_t* gate_bf = (uint16_t*)(ws + (1u << 20));
  uint16_t* up_bf   = gate_bf + (size_t)NE * D_HID * HE;
  uint16_t* down_bf = up_bf   + (size_t)NE * D_HID * HE;

  (void)hipFuncSetAttribute((const void*)moe_kernel,
                            hipFuncAttributeMaxDynamicSharedMemorySize, SMEM_BYTES);

  init_counts<<<1, 32, 0, stream>>>(counts);
  route_kernel<<<T_TOK / 8, 256, 0, stream>>>(mu, tok, rw, counts, lists);
  transpose_cvt<<<NE * (D_HID / 32) * (HE / 32), 256, 0, stream>>>(gate, gate_bf, D_HID, HE);
  transpose_cvt<<<NE * (D_HID / 32) * (HE / 32), 256, 0, stream>>>(up, up_bf, D_HID, HE);
  transpose_cvt<<<NE * (HE / 32) * (D_HID / 32), 256, 0, stream>>>(down, down_bf, HE, D_HID);
  moe_kernel<<<NE * (T_TOK / TM), 256, SMEM_BYTES, stream>>>(
      hidden, counts, lists, gate_bf, up_bf, down_bf, out);
}
